// model_bertcrf_79242146611575
// MI455X (gfx1250) — matrix operand order repaired and checked by reading, not compiled, not run on hardware
//
#include <hip/hip_runtime.h>
#include <math.h>

#define SEQ 512
#define HID 768
#define NHEAD 12
#define DHEAD 64
#define FFN 3072
#define NLAY 12
#define NTAG 27
#define NEGV (-10000.0f)

typedef __attribute__((ext_vector_type(16))) __bf16        bf16x16;
typedef __attribute__((ext_vector_type(8)))  float         f32x8;
typedef __attribute__((ext_vector_type(4)))  float         f32x4;
typedef __attribute__((ext_vector_type(4)))  unsigned int  u32x4;
typedef __attribute__((ext_vector_type(2)))  unsigned int  u32x2;

union Frag { bf16x16 bf; u32x4 q[2]; };

__device__ __forceinline__ unsigned int pk2bf(float a, float b) {
  union { __bf16 h[2]; unsigned int u; } cv;
  cv.h[0] = (__bf16)a; cv.h[1] = (__bf16)b;
  return cv.u;
}
__device__ __forceinline__ unsigned short f2bf(float a) {
  union { __bf16 h; unsigned short u; } cv;
  cv.h = (__bf16)a;
  return cv.u;
}

// ---------------------------------------------------------------------------
// Fast GEMM: C[m,n] = act( alpha * sum_k A(m,k)*B(k,n) + bias[n] )
//   A(m,k) = A[m*ldaM + k]                  (k contiguous, always)
//   BMODE 0: B(k,n) = B[k*ldbO + n]         (n contiguous: weights, V)
//   BMODE 1: B(k,n) = B[k + n*ldbO]         (k contiguous: K^T for scores)
// All dims must be tile multiples: M%64==0, N%64==0, K%32==0. No bounds checks.
// bf16 WMMA, f32 accumulation. 64x64 block tile, 4 waves of 32x32.
// ---------------------------------------------------------------------------
template <int BMODE>
__global__ __launch_bounds__(128) void gemm_wmma_fast(
    const float* __restrict__ A, const float* __restrict__ B,
    const float* __restrict__ bias, float* __restrict__ C,
    int K, int ldaM, int ldbO, int ldc, float alpha, int act)
{
  __shared__ alignas(16) unsigned short As[64][40];  // [m][k] bf16 bits
  __shared__ alignas(16) unsigned short Bt[64][40];  // [n][k] bf16 bits

  const int tid  = threadIdx.x;
  const int lane = tid & 31;
  const int wave = tid >> 5;
  const int wm = wave >> 1, wn = wave & 1;
  const int lo = lane & 15, hi = lane >> 4;
  const int bm = blockIdx.y * 64;
  const int bn = blockIdx.x * 64;

  f32x8 acc[2][2] = {};

  // A loader: 64 rows x 32 k, float4 per thread per pass, 4 passes
  const int ar = tid >> 3;          // row 0..15 (+16/pass)
  const int ac = (tid & 7) * 4;     // k within tile
  const float* pa0 = A + (size_t)(bm + ar) * ldaM + ac;
  const size_t aStep = (size_t)16 * ldaM;

  // B loader setup
  const float* pb0;
  size_t bStep;
  int br, bc;
  if constexpr (BMODE == 1) {       // k-contiguous: fill Bt rows directly
    br = tid >> 3; bc = (tid & 7) * 4;            // br = n row, bc = k
    pb0 = B + (size_t)(bn + br) * ldbO + bc;
    bStep = (size_t)16 * ldbO;
  } else {                          // n-contiguous: transpose scatter
    br = tid >> 4; bc = (tid & 15) * 4;           // br = k row, bc = n
    pb0 = B + (size_t)br * ldbO + bn + bc;
    bStep = (size_t)8 * ldbO;
  }

  for (int k0 = 0; k0 < K; k0 += 32) {
    {
      const float* pa = pa0 + k0;
      #pragma unroll
      for (int i = 0; i < 4; ++i) {
        f32x4 v = *(const f32x4*)pa;
        u32x2 pk = { pk2bf(v[0], v[1]), pk2bf(v[2], v[3]) };
        *(u32x2*)&As[ar + i * 16][ac] = pk;
        pa += aStep;
      }
    }
    if constexpr (BMODE == 1) {
      const float* pb = pb0 + k0;
      #pragma unroll
      for (int i = 0; i < 4; ++i) {
        f32x4 v = *(const f32x4*)pb;
        u32x2 pk = { pk2bf(v[0], v[1]), pk2bf(v[2], v[3]) };
        *(u32x2*)&Bt[br + i * 16][bc] = pk;
        pb += bStep;
      }
    } else {
      const float* pb = pb0 + (size_t)k0 * ldbO;
      #pragma unroll
      for (int i = 0; i < 4; ++i) {
        f32x4 v = *(const f32x4*)pb;
        const int kk = br + i * 8;
        Bt[bc + 0][kk] = f2bf(v[0]);
        Bt[bc + 1][kk] = f2bf(v[1]);
        Bt[bc + 2][kk] = f2bf(v[2]);
        Bt[bc + 3][kk] = f2bf(v[3]);
        pb += bStep;
      }
    }
    __syncthreads();

    #pragma unroll
    for (int mi = 0; mi < 2; ++mi) {
      Frag a;
      const int mrow = wm * 32 + mi * 16 + lo;
      // A 16x32 bf16 layout: elem j -> K = j + (j>=8?8:0) + hi*8
      a.q[0] = *(const u32x4*)&As[mrow][hi * 8];
      a.q[1] = *(const u32x4*)&As[mrow][16 + hi * 8];
      #pragma unroll
      for (int ni = 0; ni < 2; ++ni) {
        Frag b;
        const int ncol = wn * 32 + ni * 16 + lo;
        // B 32x16 bf16 layout: elem j -> K = j + (j>=8?8:0) + hi*8, N = lane&15
        b.q[0] = *(const u32x4*)&Bt[ncol][hi * 8];
        b.q[1] = *(const u32x4*)&Bt[ncol][16 + hi * 8];
        acc[mi][ni] = __builtin_amdgcn_wmma_f32_16x16x32_bf16(
            false, a.bf, false, b.bf, (short)0, acc[mi][ni], false, false);
      }
    }
    __syncthreads();
  }

  // epilogue: C/D layout -> VGPR g holds M = g + hi*8, N = lane&15
  #pragma unroll
  for (int mi = 0; mi < 2; ++mi) {
    #pragma unroll
    for (int ni = 0; ni < 2; ++ni) {
      const int ncol = bn + wn * 32 + ni * 16 + lo;
      const float bv = bias ? bias[ncol] : 0.0f;
      float* pc = C + (size_t)(bm + wm * 32 + mi * 16 + hi * 8) * ldc + ncol;
      #pragma unroll
      for (int g = 0; g < 8; ++g) {
        float v = alpha * acc[mi][ni][g] + bv;
        if (act) v = 0.5f * v * (1.0f + erff(v * 0.70710678118654752f)); // exact GELU
        *pc = v;
        pc += ldc;
      }
    }
  }
}

// ---------------------------------------------------------------------------
__device__ __forceinline__ void block_sum2(float& a, float& b) {
  __shared__ float ra[8], rb[8];
  const int lane = threadIdx.x & 31, w = threadIdx.x >> 5;
  #pragma unroll
  for (int off = 16; off > 0; off >>= 1) {
    a += __shfl_xor(a, off, 32);
    b += __shfl_xor(b, off, 32);
  }
  if (lane == 0) { ra[w] = a; rb[w] = b; }
  __syncthreads();
  a = 0.0f; b = 0.0f;
  #pragma unroll
  for (int i = 0; i < 8; ++i) { a += ra[i]; b += rb[i]; }
  __syncthreads();
}

// h[t] = LN(word_emb[sent[t]] + pos_emb[t] + type_emb[0]) * gs + gb
__global__ __launch_bounds__(256) void embed_ln_k(
    const int* __restrict__ sent, const float* __restrict__ we,
    const float* __restrict__ pe, const float* __restrict__ te,
    const float* __restrict__ gs, const float* __restrict__ gb,
    float* __restrict__ h)
{
  const int t = blockIdx.x, tid = threadIdx.x;
  const int wid = sent[t];
  float x[3]; float s = 0.0f, q = 0.0f;
  #pragma unroll
  for (int i = 0; i < 3; ++i) {
    int c = tid + i * 256;
    float v = we[(size_t)wid * HID + c] + pe[(size_t)t * HID + c] + te[c];
    x[i] = v; s += v; q += v * v;
  }
  block_sum2(s, q);
  const float mean = s * (1.0f / HID);
  const float var  = q * (1.0f / HID) - mean * mean;   // biased var (jnp.var)
  const float r = rsqrtf(var + 1e-12f);
  #pragma unroll
  for (int i = 0; i < 3; ++i) {
    int c = tid + i * 256;
    h[(size_t)t * HID + c] = (x[i] - mean) * r * gs[c] + gb[c];
  }
}

// h[t] = LN(h[t] + o[t]) * gs + gb   (in place)
__global__ __launch_bounds__(256) void residual_ln_k(
    float* __restrict__ h, const float* __restrict__ o,
    const float* __restrict__ gs, const float* __restrict__ gb)
{
  const int t = blockIdx.x, tid = threadIdx.x;
  float x[3]; float s = 0.0f, q = 0.0f;
  #pragma unroll
  for (int i = 0; i < 3; ++i) {
    int c = tid + i * 256;
    float v = h[(size_t)t * HID + c] + o[(size_t)t * HID + c];
    x[i] = v; s += v; q += v * v;
  }
  block_sum2(s, q);
  const float mean = s * (1.0f / HID);
  const float var  = q * (1.0f / HID) - mean * mean;
  const float r = rsqrtf(var + 1e-12f);
  #pragma unroll
  for (int i = 0; i < 3; ++i) {
    int c = tid + i * 256;
    h[(size_t)t * HID + c] = (x[i] - mean) * r * gs[c] + gb[c];
  }
}

// row softmax over last dim (512); grid = (SEQ rows, NHEAD)
__global__ __launch_bounds__(256) void softmax_k(float* __restrict__ sc) {
  const size_t row = (size_t)blockIdx.y * SEQ + blockIdx.x;
  float* p = sc + row * SEQ;
  const int tid = threadIdx.x, lane = tid & 31, w = tid >> 5;
  __shared__ float red[8];
  float v0 = p[tid], v1 = p[tid + 256];
  float m = fmaxf(v0, v1);
  #pragma unroll
  for (int off = 16; off > 0; off >>= 1) m = fmaxf(m, __shfl_xor(m, off, 32));
  if (lane == 0) red[w] = m;
  __syncthreads();
  float bm = red[0];
  #pragma unroll
  for (int i = 1; i < 8; ++i) bm = fmaxf(bm, red[i]);
  __syncthreads();
  float e0 = expf(v0 - bm), e1 = expf(v1 - bm);
  float s = e0 + e1;
  #pragma unroll
  for (int off = 16; off > 0; off >>= 1) s += __shfl_xor(s, off, 32);
  if (lane == 0) red[w] = s;
  __syncthreads();
  float tot = 0.0f;
  #pragma unroll
  for (int i = 0; i < 8; ++i) tot += red[i];
  const float inv = 1.0f / tot;
  p[tid] = e0 * inv; p[tid + 256] = e1 * inv;
}

__global__ void attn_bias_k(const int* __restrict__ mask, float* __restrict__ ab) {
  int n = threadIdx.x;
  if (n < SEQ) ab[n] = (1.0f - (float)mask[n]) * NEGV;
}

// classifier head: logits[t, j] = h[t,:] . cls_W[:, j] + cls_b[j]   (N=27, tiny)
__global__ __launch_bounds__(256) void cls_k(
    const float* __restrict__ h, const float* __restrict__ W,
    const float* __restrict__ b, float* __restrict__ logits)
{
  __shared__ float hs[HID];
  const int t = blockIdx.x, tid = threadIdx.x;
  for (int c = tid; c < HID; c += 256) hs[c] = h[(size_t)t * HID + c];
  __syncthreads();
  if (tid < NTAG) {
    float s = b[tid];
    for (int c = 0; c < HID; ++c) s += hs[c] * W[(size_t)c * NTAG + tid];
    logits[(size_t)t * NTAG + tid] = s;
  }
}

// Single-wave CRF Viterbi. out = [path_score, best_path(511 values as float)]
__global__ void viterbi_k(const float* __restrict__ logits, const int* __restrict__ mask,
                          const float* __restrict__ trans, float* __restrict__ out,
                          int* __restrict__ bp)
{
  const int lane = threadIdx.x;
  __shared__ float fv[32];
  __shared__ float tsh[NTAG * NTAG];
  __shared__ int Ls;
  for (int i = lane; i < NTAG * NTAG; i += 32) tsh[i] = trans[i];
  if (lane == 0) { int L = 0; for (int i = 0; i < SEQ; ++i) L += mask[i]; Ls = L; }
  fv[lane] = (lane == 25) ? 0.0f : NEGV;
  __syncthreads();
  const int L = Ls;

  for (int t = 1; t <= SEQ - 2; ++t) {
    float best = -3.0e38f; int bi = 0;
    if (lane < NTAG) {
      for (int i = 0; i < NTAG; ++i) {
        float v = fv[i] + tsh[i * NTAG + lane];
        if (v > best) { best = v; bi = i; }     // first max (argmax semantics)
      }
    }
    __syncthreads();
    if (lane < NTAG) {
      const bool valid = t < (L - 1);
      if (valid) fv[lane] = best + logits[(size_t)t * NTAG + lane];
      bp[(t - 1) * NTAG + lane] = valid ? bi : lane;
    }
    __syncthreads();
  }

  if (lane == 0) {
    int best = 0; float bestv = fv[0] + tsh[0 * NTAG + 26];
    for (int j = 1; j < NTAG; ++j) {
      float v = fv[j] + tsh[j * NTAG + 26];
      if (v > bestv) { bestv = v; best = j; }
    }
    out[0] = bestv;
    out[1 + (SEQ - 2)] = (float)best;          // best_path[510]
    int b = best;
    for (int idx = SEQ - 3; idx >= 0; --idx) { // reverse backtrace
      b = bp[idx * NTAG + b];
      out[1 + idx] = (float)b;
    }
  }
}

// ---------------------------------------------------------------------------
static inline void gemm_nc(hipStream_t st, const float* A, const float* B,
                           const float* bias, float* C, int M, int N, int K,
                           int ldaM, int ldbK, int ldc, float alpha, int act) {
  dim3 grid(N / 64, M / 64);
  gemm_wmma_fast<0><<<grid, dim3(128), 0, st>>>(A, B, bias, C, K, ldaM, ldbK, ldc, alpha, act);
}
static inline void gemm_kc(hipStream_t st, const float* A, const float* B,
                           const float* bias, float* C, int M, int N, int K,
                           int ldaM, int ldbN, int ldc, float alpha, int act) {
  dim3 grid(N / 64, M / 64);
  gemm_wmma_fast<1><<<grid, dim3(128), 0, st>>>(A, B, bias, C, K, ldaM, ldbN, ldc, alpha, act);
}

extern "C" void kernel_launch(void* const* d_in, const int* in_sizes, int n_in,
                              void* d_out, int out_size, void* d_ws, size_t ws_size,
                              hipStream_t stream) {
  const int*   sent   = (const int*)d_in[0];
  const int*   mask   = (const int*)d_in[1];
  const float* word_e = (const float*)d_in[2];
  const float* pos_e  = (const float*)d_in[3];
  const float* type_e = (const float*)d_in[4];
  const float* eln_s  = (const float*)d_in[5];
  const float* eln_b  = (const float*)d_in[6];
  const float* Wq     = (const float*)d_in[7];
  const float* bq     = (const float*)d_in[8];
  const float* Wk     = (const float*)d_in[9];
  const float* bk     = (const float*)d_in[10];
  const float* Wv     = (const float*)d_in[11];
  const float* bv     = (const float*)d_in[12];
  const float* Wo     = (const float*)d_in[13];
  const float* bo     = (const float*)d_in[14];
  const float* ln1s   = (const float*)d_in[15];
  const float* ln1b   = (const float*)d_in[16];
  const float* W1     = (const float*)d_in[17];
  const float* b1     = (const float*)d_in[18];
  const float* W2     = (const float*)d_in[19];
  const float* b2     = (const float*)d_in[20];
  const float* ln2s   = (const float*)d_in[21];
  const float* ln2b   = (const float*)d_in[22];
  const float* clsW   = (const float*)d_in[23];
  const float* clsb   = (const float*)d_in[24];
  const float* trans  = (const float*)d_in[25];

  // workspace layout (floats)
  float* ws     = (float*)d_ws;
  float* h      = ws;                                  // 512*768
  float* q      = h    + (size_t)SEQ * HID;            // 512*768
  float* kb     = q    + (size_t)SEQ * HID;            // 512*768
  float* vb     = kb   + (size_t)SEQ * HID;            // 512*768
  float* ctx    = vb   + (size_t)SEQ * HID;            // 512*768
  float* ff     = ctx  + (size_t)SEQ * HID;            // 512*3072
  float* scores = ff   + (size_t)SEQ * FFN;            // 12*512*512
  float* abias  = scores + (size_t)NHEAD * SEQ * SEQ;  // 512
  float* logits = abias + SEQ;                         // 512*27
  int*   bp     = (int*)(logits + (size_t)SEQ * NTAG); // 510*27 ints

  attn_bias_k<<<dim3(1), dim3(SEQ), 0, stream>>>(mask, abias);
  embed_ln_k<<<dim3(SEQ), dim3(256), 0, stream>>>(sent, word_e, pos_e, type_e,
                                                  eln_s, eln_b, h);

  const float scale = 0.125f;  // 1/sqrt(64)
  for (int l = 0; l < NLAY; ++l) {
    const size_t wHH = (size_t)l * HID * HID;
    const size_t wHF = (size_t)l * HID * FFN;
    const size_t vH  = (size_t)l * HID;
    const size_t vF  = (size_t)l * FFN;

    // Q,K,V projections (weights: n-contiguous, ldbK = HID)
    gemm_nc(stream, h, Wq + wHH, bq + vH, q,  SEQ, HID, HID, HID, HID, HID, 1.0f, 0);
    gemm_nc(stream, h, Wk + wHH, bk + vH, kb, SEQ, HID, HID, HID, HID, HID, 1.0f, 0);
    gemm_nc(stream, h, Wv + wHH, bv + vH, vb, SEQ, HID, HID, HID, HID, HID, 1.0f, 0);

    // per-head scores: sc = scale * q_h @ k_h^T + attn_bias  (B k-contiguous, ldbN = HID)
    for (int hh = 0; hh < NHEAD; ++hh)
      gemm_kc(stream, q + hh * DHEAD, kb + hh * DHEAD, abias,
              scores + (size_t)hh * SEQ * SEQ,
              SEQ, SEQ, DHEAD, HID, HID, SEQ, scale, 0);

    softmax_k<<<dim3(SEQ, NHEAD), dim3(256), 0, stream>>>(scores);

    // ctx_h = a_h @ v_h  (B n-contiguous, ldbK = HID)
    for (int hh = 0; hh < NHEAD; ++hh)
      gemm_nc(stream, scores + (size_t)hh * SEQ * SEQ, vb + hh * DHEAD, nullptr,
              ctx + hh * DHEAD, SEQ, DHEAD, SEQ, SEQ, HID, HID, 1.0f, 0);

    // output projection + residual LN
    gemm_nc(stream, ctx, Wo + wHH, bo + vH, ff, SEQ, HID, HID, HID, HID, HID, 1.0f, 0);
    residual_ln_k<<<dim3(SEQ), dim3(256), 0, stream>>>(h, ff, ln1s + vH, ln1b + vH);

    // FFN
    gemm_nc(stream, h,  W1 + wHF, b1 + vF, ff,  SEQ, FFN, HID, HID, FFN, FFN, 1.0f, 1);
    gemm_nc(stream, ff, W2 + wHF, b2 + vH, ctx, SEQ, HID, FFN, FFN, HID, HID, 1.0f, 0);
    residual_ln_k<<<dim3(SEQ), dim3(256), 0, stream>>>(h, ctx, ln2s + vH, ln2b + vH);
  }

  // classifier head + CRF Viterbi
  cls_k<<<dim3(SEQ), dim3(256), 0, stream>>>(h, clsW, clsb, logits);
  viterbi_k<<<dim3(1), dim3(32), 0, stream>>>(logits, mask, trans, (float*)d_out, bp);
}
